// LSTMQ_dynamic_52664888983785
// MI455X (gfx1250) — 16-bit and 8-bit matrix operand order repaired and checked by reading, compiled, not run on hardware
//
#include <hip/hip_runtime.h>
#include <cstdint>

typedef __bf16 bhalf;
typedef __attribute__((ext_vector_type(16))) __bf16 v16bf;
typedef __attribute__((ext_vector_type(8)))  float  v8f;
typedef __attribute__((ext_vector_type(8)))  int    v8i;

struct alignas(16) U4 { unsigned x, y, z, w; };
struct alignas(8)  U2 { unsigned x, y; };
union ABf { v16bf v; U4 q[2]; };
union AI8 { v8i  v; U2 d2[4]; U4 d4[2]; };

#define TSEQ 512
#define BB_  32
#define CC_  512
#define HH_  1024
#define FH_  4096
#define NWG  64
#define EPSQ 1e-8f
#define FINF 3.402823466e38f

#define CELL_WIH 0
#define CELL_WHH 1
#define CELL_Y   2
#define CELL_IH(t)     (16 + (t))
#define CELL_STEP(t,s) (528 + (t)*12 + (s))

static constexpr size_t O_MAXC = 0;
static constexpr size_t O_CNT  = 65536;
static constexpr size_t O_QH   = 65792;
static constexpr size_t O_CST  = 98560;
static constexpr size_t ZSIZE  = 229632;
static constexpr size_t O_MINC = 229632;
static constexpr size_t O_WIHB = 295168;
static constexpr size_t O_XB   = O_WIHB + (size_t)FH_*CC_*2;
static constexpr size_t O_WHHQ = O_XB + (size_t)BB_*TSEQ*CC_*2;
static constexpr size_t O_RS   = O_WHHQ + (size_t)FH_*HH_;
static constexpr size_t O_IHQ  = O_RS + 16384;
static constexpr size_t O_B4H  = O_IHQ + (size_t)TSEQ*BB_*FH_;
static constexpr size_t O_BA   = O_B4H + (size_t)BB_*FH_*4;
static constexpr size_t O_BB2  = O_BA + (size_t)BB_*HH_*4;
static constexpr size_t WS_NEED = O_BB2 + (size_t)BB_*HH_*4;

__device__ __forceinline__ unsigned fenc(float f) {
  unsigned u = __float_as_uint(f);
  return (u & 0x80000000u) ? ~u : (u | 0x80000000u);
}
__device__ __forceinline__ float fdec(unsigned e) {
  unsigned u = (e & 0x80000000u) ? (e & 0x7fffffffu) : ~e;
  return __uint_as_float(u);
}
__device__ __forceinline__ unsigned ldcell(const unsigned* p) {
  return __hip_atomic_load(p, __ATOMIC_RELAXED, __HIP_MEMORY_SCOPE_AGENT);
}
__device__ __forceinline__ float2 get_sm(const unsigned* minc, const unsigned* maxc, int cell) {
  float mn = fdec(ldcell(minc + cell));
  float mx = fdec(ldcell(maxc + cell));
  float s  = fmaxf((mx - mn) * (1.0f / 255.0f), EPSQ);
  return make_float2(s, mn);
}
__device__ __forceinline__ float fq_apply(float v, float s, float mn) {
  float q = rintf((v - mn) / s);
  q = fminf(fmaxf(q, 0.0f), 255.0f);
  return fmaf(q, s, mn);
}
__device__ __forceinline__ void push_minmax(float lmin, float lmax,
                                            unsigned* minc, unsigned* maxc, int cell) {
  #pragma unroll
  for (int m = 16; m; m >>= 1) {
    lmin = fminf(lmin, __shfl_xor(lmin, m, 32));
    lmax = fmaxf(lmax, __shfl_xor(lmax, m, 32));
  }
  if ((threadIdx.x & 31) == 0) {
    atomicMin(minc + cell, fenc(lmin));
    atomicMax(maxc + cell, fenc(lmax));
  }
}
__device__ __forceinline__ void gbar(unsigned* cnt, unsigned* sph) {
  __threadfence();
  __syncthreads();
  if (threadIdx.x == 0) {
    unsigned ph = *sph + 1;
    *sph = ph;
    atomicAdd(cnt, 1u);
    unsigned target = ph * NWG;
    while (__hip_atomic_load(cnt, __ATOMIC_RELAXED, __HIP_MEMORY_SCOPE_AGENT) < target)
      __builtin_amdgcn_s_sleep(2);
    __threadfence();
  }
  __syncthreads();
}

__global__ void __launch_bounds__(256) k_wabsmax(const float* __restrict__ Wih,
                                                 const float* __restrict__ Whh,
                                                 unsigned* maxc) {
  size_t stride = (size_t)gridDim.x * blockDim.x;
  size_t tid = (size_t)blockIdx.x * blockDim.x + threadIdx.x;
  float m1 = 0.0f;
  for (size_t i = tid; i < (size_t)FH_ * CC_; i += stride) m1 = fmaxf(m1, fabsf(Wih[i]));
  #pragma unroll
  for (int m = 16; m; m >>= 1) m1 = fmaxf(m1, __shfl_xor(m1, m, 32));
  if ((threadIdx.x & 31) == 0) atomicMax(maxc + CELL_WIH, __float_as_uint(m1));
  float m2 = 0.0f;
  for (size_t i = tid; i < (size_t)FH_ * HH_; i += stride) m2 = fmaxf(m2, fabsf(Whh[i]));
  #pragma unroll
  for (int m = 16; m; m >>= 1) m2 = fmaxf(m2, __shfl_xor(m2, m, 32));
  if ((threadIdx.x & 31) == 0) atomicMax(maxc + CELL_WHH, __float_as_uint(m2));
}

__global__ void __launch_bounds__(256) k_xcvt(const float* __restrict__ x, bhalf* __restrict__ xb) {
  size_t i = (size_t)blockIdx.x * blockDim.x + threadIdx.x;
  xb[i] = (bhalf)x[i];
}

__global__ void __launch_bounds__(256) k_wq_ih(const float* __restrict__ W, bhalf* __restrict__ out,
                                               const unsigned* maxc) {
  size_t i = (size_t)blockIdx.x * blockDim.x + threadIdx.x;
  float s = fmaxf(__uint_as_float(ldcell(maxc + CELL_WIH)) * (1.0f / 127.0f), EPSQ);
  float q = fminf(fmaxf(rintf(W[i] / s), -128.0f), 127.0f);
  out[i] = (bhalf)(q * s);
}

__global__ void __launch_bounds__(256) k_wq_hh(const float* __restrict__ W, signed char* __restrict__ out,
                                               int* __restrict__ rowsum, const unsigned* maxc) {
  int j = blockIdx.x;
  float s = fmaxf(__uint_as_float(ldcell(maxc + CELL_WHH)) * (1.0f / 127.0f), EPSQ);
  int local = 0;
  for (int k = threadIdx.x; k < HH_; k += 256) {
    float q = fminf(fmaxf(rintf(W[(size_t)j * HH_ + k] / s), -128.0f), 127.0f);
    out[(size_t)j * HH_ + k] = (signed char)q;
    local += (int)q;
  }
  #pragma unroll
  for (int m = 16; m; m >>= 1) local += __shfl_xor(local, m, 32);
  __shared__ int sh[8];
  if ((threadIdx.x & 31) == 0) sh[threadIdx.x >> 5] = local;
  __syncthreads();
  if (threadIdx.x == 0) {
    int t = 0;
    #pragma unroll
    for (int w = 0; w < 8; w++) t += sh[w];
    rowsum[j] = t;
  }
}

__global__ void __launch_bounds__(256) k_ih_gemm(const bhalf* __restrict__ xb,
                                                 const bhalf* __restrict__ wb,
                                                 const float* __restrict__ bih,
                                                 unsigned char* __restrict__ ihq,
                                                 unsigned* minc, unsigned* maxc, int do_quant) {
  int lane = threadIdx.x & 31;
  int tile = blockIdx.x * 8 + (threadIdx.x >> 5);
  int nt = tile & 255, mt = tile >> 8;
  int m0 = mt * 16, j0 = nt * 16;
  int hlf = lane >> 4, lm = lane & 15;
  int m = m0 + lm, b = m & 31, tt = m >> 5;
  const bhalf* arow = xb + ((size_t)b * TSEQ + (size_t)tt) * CC_;
  const bhalf* brow = wb + (size_t)(j0 + lm) * CC_;
  v8f acc = {};
  #pragma unroll
  for (int k0 = 0; k0 < CC_; k0 += 32) {
    __builtin_prefetch(brow + k0 + 128, 0, 3);
    ABf Af, Bf;
    const U4* pa = (const U4*)(arow + k0 + hlf * 8);
    Af.q[0] = pa[0]; Af.q[1] = pa[2];
    const U4* pb = (const U4*)(brow + k0 + hlf * 8);
    Bf.q[0] = pb[0]; Bf.q[1] = pb[2];
    acc = __builtin_amdgcn_wmma_f32_16x16x32_bf16(false, Af.v, false, Bf.v,
                                                  (short)0, acc, false, false);
  }
  int step = mt >> 1;
  float bv = bih[j0 + lm];
  if (!do_quant) {
    float lmin = FINF, lmax = -FINF;
    #pragma unroll
    for (int r = 0; r < 8; r++) {
      float v = acc[r] + bv;
      lmin = fminf(lmin, v); lmax = fmaxf(lmax, v);
    }
    push_minmax(lmin, lmax, minc, maxc, CELL_IH(step));
  } else {
    float2 sm = get_sm(minc, maxc, CELL_IH(step));
    #pragma unroll
    for (int r = 0; r < 8; r++) {
      float v = acc[r] + bv;
      float q = fminf(fmaxf(rintf((v - sm.y) / sm.x), 0.0f), 255.0f);
      ihq[(size_t)(m0 + r + 8 * hlf) * FH_ + j0 + lm] = (unsigned char)q;
    }
  }
}

__global__ void __launch_bounds__(256, 1) k_lstm_scan(
    const unsigned char* __restrict__ ihq, const signed char* __restrict__ whhq,
    const int* __restrict__ rowsum, const float* __restrict__ bhh,
    unsigned char* __restrict__ qh, float* __restrict__ cstate,
    float* __restrict__ buf4h, float* __restrict__ bufA, float* __restrict__ bufB,
    unsigned* minc, unsigned* maxc, unsigned* cnt, float* __restrict__ out) {
  __shared__ unsigned s_ph;
  if (threadIdx.x == 0) s_ph = 0;
  __syncthreads();

  const int lane = threadIdx.x & 31;
  const int hlf = lane >> 4, lm = lane & 15;
  const int wid = (blockIdx.x << 3) + (threadIdx.x >> 5);
  const int tidg = blockIdx.x * 256 + threadIdx.x;
  const int m0 = (wid >> 8) * 16, j0 = (wid & 255) * 16;
  const int jcol = j0 + lm;
  const float ws_hh = fmaxf(__uint_as_float(ldcell(maxc + CELL_WHH)) * (1.0f / 127.0f), EPSQ);
  const float rs = (float)rowsum[jcol];
  const float bb = bhh[jcol];
  const int qd = (tidg & 4095) >> 10;

  for (int t = 0; t < TSEQ; ++t) {
    {
      float hs = 1.0f, hmn = 0.0f;
      if (t > 0) { float2 sm = get_sm(minc, maxc, CELL_STEP(t - 1, 10)); hs = sm.x; hmn = sm.y; }
      const unsigned char* arow = qh + (m0 + lm) * HH_;
      const signed char*   brow = whhq + (size_t)jcol * HH_;
      v8i acc = {};
      #pragma unroll
      for (int kc = 0; kc < 16; kc++) {
        int k0 = kc * 64;
        AI8 Af, Bf;
        const U2* pa = (const U2*)(arow + k0 + hlf * 8);
        Af.d2[0] = pa[0]; Af.d2[1] = pa[2]; Af.d2[2] = pa[4]; Af.d2[3] = pa[6];
        const U2* pb = (const U2*)(brow + k0 + hlf * 8);
        Bf.d2[0] = pb[0]; Bf.d2[1] = pb[2]; Bf.d2[2] = pb[4]; Bf.d2[3] = pb[6];
        acc = __builtin_amdgcn_wmma_i32_16x16x64_iu8(false, Af.v, true, Bf.v, acc, false, false);
      }
      float base = hmn * ws_hh * rs + bb;
      float lmin = FINF, lmax = -FINF;
      #pragma unroll
      for (int r = 0; r < 8; r++) {
        float v = fmaf(hs * ws_hh, (float)acc[r], base);
        buf4h[(m0 + r + 8 * hlf) * FH_ + jcol] = v;
        lmin = fminf(lmin, v); lmax = fmaxf(lmax, v);
      }
      push_minmax(lmin, lmax, minc, maxc, CELL_STEP(t, 0));
    }
    gbar(cnt, &s_ph);
    {
      float2 s0 = get_sm(minc, maxc, CELL_STEP(t, 0));
      float2 si = get_sm(minc, maxc, CELL_IH(t));
      float lmin = FINF, lmax = -FINF;
      #pragma unroll
      for (int e = 0; e < 8; e++) {
        int idx = tidg + e * 16384;
        int b = idx >> 12, j = idx & 4095;
        float hhq = fq_apply(buf4h[idx], s0.x, s0.y);
        float ihv = fmaf((float)ihq[(size_t)(t * BB_ + b) * FH_ + j], si.x, si.y);
        float g = ihv + hhq;
        buf4h[idx] = g;
        lmin = fminf(lmin, g); lmax = fmaxf(lmax, g);
      }
      push_minmax(lmin, lmax, minc, maxc, CELL_STEP(t, 1));
    }
    gbar(cnt, &s_ph);
    {
      float2 s1 = get_sm(minc, maxc, CELL_STEP(t, 1));
      float lmin = FINF, lmax = -FINF;
      #pragma unroll
      for (int e = 0; e < 8; e++) {
        int idx = tidg + e * 16384;
        float g = fq_apply(buf4h[idx], s1.x, s1.y);
        float act = (qd == 2) ? tanhf(g) : (1.0f / (1.0f + expf(-g)));
        buf4h[idx] = act;
        lmin = fminf(lmin, act); lmax = fmaxf(lmax, act);
      }
      push_minmax(lmin, lmax, minc, maxc, CELL_STEP(t, 2 + qd));
    }
    gbar(cnt, &s_ph);
    {
      float2 s2 = get_sm(minc, maxc, CELL_STEP(t, 2));
      float2 s3 = get_sm(minc, maxc, CELL_STEP(t, 3));
      float2 s4 = get_sm(minc, maxc, CELL_STEP(t, 4));
      float amin = FINF, amax = -FINF, bmin = FINF, bmax = -FINF;
      #pragma unroll
      for (int e = 0; e < 2; e++) {
        int idx = tidg + e * 16384;
        int b = idx >> 10, j = idx & 1023;
        int base = b * FH_ + j;
        float gi = fq_apply(buf4h[base],        s2.x, s2.y);
        float gf = fq_apply(buf4h[base + 1024], s3.x, s3.y);
        float gg = fq_apply(buf4h[base + 2048], s4.x, s4.y);
        float av = gf * cstate[idx];
        float bv = gi * gg;
        bufA[idx] = av; bufB[idx] = bv;
        amin = fminf(amin, av); amax = fmaxf(amax, av);
        bmin = fminf(bmin, bv); bmax = fmaxf(bmax, bv);
      }
      push_minmax(amin, amax, minc, maxc, CELL_STEP(t, 6));
      push_minmax(bmin, bmax, minc, maxc, CELL_STEP(t, 7));
    }
    gbar(cnt, &s_ph);
    {
      float2 s6 = get_sm(minc, maxc, CELL_STEP(t, 6));
      float2 s7 = get_sm(minc, maxc, CELL_STEP(t, 7));
      float lmin = FINF, lmax = -FINF;
      #pragma unroll
      for (int e = 0; e < 2; e++) {
        int idx = tidg + e * 16384;
        float cn = fq_apply(bufA[idx], s6.x, s6.y) + fq_apply(bufB[idx], s7.x, s7.y);
        bufA[idx] = cn;
        lmin = fminf(lmin, cn); lmax = fmaxf(lmax, cn);
      }
      push_minmax(lmin, lmax, minc, maxc, CELL_STEP(t, 8));
    }
    gbar(cnt, &s_ph);
    {
      float2 s8 = get_sm(minc, maxc, CELL_STEP(t, 8));
      float lmin = FINF, lmax = -FINF;
      #pragma unroll
      for (int e = 0; e < 2; e++) {
        int idx = tidg + e * 16384;
        float cq = fq_apply(bufA[idx], s8.x, s8.y);
        cstate[idx] = cq;
        float th = tanhf(cq);
        bufB[idx] = th;
        lmin = fminf(lmin, th); lmax = fmaxf(lmax, th);
      }
      push_minmax(lmin, lmax, minc, maxc, CELL_STEP(t, 9));
    }
    gbar(cnt, &s_ph);
    {
      float2 s5 = get_sm(minc, maxc, CELL_STEP(t, 5));
      float2 s9 = get_sm(minc, maxc, CELL_STEP(t, 9));
      float lmin = FINF, lmax = -FINF;
      #pragma unroll
      for (int e = 0; e < 2; e++) {
        int idx = tidg + e * 16384;
        int b = idx >> 10, j = idx & 1023;
        float go_ = fq_apply(buf4h[b * FH_ + 3072 + j], s5.x, s5.y);
        float thq = fq_apply(bufB[idx], s9.x, s9.y);
        float hv = go_ * thq;
        bufA[idx] = hv;
        lmin = fminf(lmin, hv); lmax = fmaxf(lmax, hv);
      }
      push_minmax(lmin, lmax, minc, maxc, CELL_STEP(t, 10));
    }
    gbar(cnt, &s_ph);
    {
      float2 sh_ = get_sm(minc, maxc, CELL_STEP(t, 10));
      #pragma unroll
      for (int e = 0; e < 2; e++) {
        int idx = tidg + e * 16384;
        int b = idx >> 10, j = idx & 1023;
        float q = fminf(fmaxf(rintf((bufA[idx] - sh_.y) / sh_.x), 0.0f), 255.0f);
        qh[idx] = (unsigned char)q;
        out[(size_t)b * (TSEQ * HH_) + (size_t)t * HH_ + j] = fmaf(q, sh_.x, sh_.y);
      }
    }
    gbar(cnt, &s_ph);
  }
}

__global__ void __launch_bounds__(256) k_yminmax(const float* __restrict__ out,
                                                 unsigned* minc, unsigned* maxc) {
  size_t stride = (size_t)gridDim.x * blockDim.x;
  size_t tid = (size_t)blockIdx.x * blockDim.x + threadIdx.x;
  float lmin = FINF, lmax = -FINF;
  for (size_t i = tid; i < (size_t)BB_ * TSEQ * HH_; i += stride) {
    float v = out[i];
    lmin = fminf(lmin, v); lmax = fmaxf(lmax, v);
  }
  push_minmax(lmin, lmax, minc, maxc, CELL_Y);
}

__global__ void __launch_bounds__(256) k_yapply(float* __restrict__ out,
                                                const unsigned* minc, const unsigned* maxc) {
  float2 sm = get_sm(minc, maxc, CELL_Y);
  size_t i = (size_t)blockIdx.x * blockDim.x + threadIdx.x;
  out[i] = fq_apply(out[i], sm.x, sm.y);
}

extern "C" void kernel_launch(void* const* d_in, const int* in_sizes, int n_in,
                              void* d_out, int out_size, void* d_ws, size_t ws_size,
                              hipStream_t stream) {
  (void)in_sizes; (void)n_in; (void)out_size;
  if (ws_size < WS_NEED) return;

  const float* x   = (const float*)d_in[0];
  const float* Wih = (const float*)d_in[1];
  const float* Whh = (const float*)d_in[2];
  const float* bih = (const float*)d_in[3];
  const float* bhh = (const float*)d_in[4];
  float* out = (float*)d_out;
  char* ws = (char*)d_ws;

  unsigned*      maxc   = (unsigned*)(ws + O_MAXC);
  unsigned*      cnt    = (unsigned*)(ws + O_CNT);
  unsigned char* qh     = (unsigned char*)(ws + O_QH);
  float*         cstate = (float*)(ws + O_CST);
  unsigned*      minc   = (unsigned*)(ws + O_MINC);
  bhalf*         wihb   = (bhalf*)(ws + O_WIHB);
  bhalf*         xb     = (bhalf*)(ws + O_XB);
  signed char*   whhq   = (signed char*)(ws + O_WHHQ);
  int*           rowsum = (int*)(ws + O_RS);
  unsigned char* ihq    = (unsigned char*)(ws + O_IHQ);
  float*         buf4h  = (float*)(ws + O_B4H);
  float*         bufA   = (float*)(ws + O_BA);
  float*         bufB   = (float*)(ws + O_BB2);

  hipMemsetAsync(ws, 0, ZSIZE, stream);
  hipMemsetAsync(ws + O_MINC, 0xFF, 65536, stream);

  k_wabsmax<<<512, 256, 0, stream>>>(Wih, Whh, maxc);
  k_xcvt<<<(BB_ * TSEQ * CC_) / 256, 256, 0, stream>>>(x, xb);
  k_wq_ih<<<(FH_ * CC_) / 256, 256, 0, stream>>>(Wih, wihb, maxc);
  k_wq_hh<<<FH_, 256, 0, stream>>>(Whh, whhq, rowsum, maxc);

  k_ih_gemm<<<32768, 256, 0, stream>>>(xb, wihb, bih, ihq, minc, maxc, 0);
  k_ih_gemm<<<32768, 256, 0, stream>>>(xb, wihb, bih, ihq, minc, maxc, 1);

  k_lstm_scan<<<NWG, 256, 0, stream>>>(ihq, whhq, rowsum, bhh, qh, cstate,
                                       buf4h, bufA, bufB, minc, maxc, cnt, out);

  k_yminmax<<<2048, 256, 0, stream>>>(out, minc, maxc);
  k_yapply<<<(BB_ * TSEQ * HH_) / 256, 256, 0, stream>>>(out, minc, maxc);
}
